// Attention_17480516895444
// MI455X (gfx1250) — matrix operand order repaired and checked by reading, not compiled, not run on hardware
//
#include <hip/hip_runtime.h>
#include <hip/hip_bf16.h>
#include <math.h>

// ---------------------------------------------------------------------------
// MI455X (gfx1250) attention: out = (Q K^T * scale) @ V - lse * colsum(V),
// then out @ w_out + b_out + x.   (log_softmax => no exp applied before @V)
// All GEMMs via v_wmma_f32_16x16x32_bf16, fp32 accumulate, wave32.
// V is produced pre-transposed (vt[b,h,d,n]); attention hot loop uses
// double-buffered global_load_async_to_lds_b128 tile copies.
// ---------------------------------------------------------------------------

typedef __bf16 bf16;
typedef __attribute__((ext_vector_type(16))) __bf16 v16bf;
typedef __attribute__((ext_vector_type(8)))  float  v8f;
typedef __attribute__((ext_vector_type(4)))  int    v4i;

#define SEQ   4096
#define DHEAD 64
#define NBH   16      // batch(2) * heads(8)
#define QK_SCALE 0.125f

#ifndef __has_builtin
#define __has_builtin(x) 0
#endif

#if __has_builtin(__builtin_amdgcn_global_load_async_to_lds_b128)
#define ASYNC_LDS 1
#else
#define ASYNC_LDS 0
#endif

#if ASYNC_LDS
#if __has_builtin(__builtin_amdgcn_s_wait_asynccnt)
#define WAIT_ASYNC(n) __builtin_amdgcn_s_wait_asynccnt(n)
#else
#define WAIT_ASYNC(n) asm volatile("s_wait_asynccnt %0" ::"i"(n) : "memory")
#endif
#else
#define WAIT_ASYNC(n) ((void)0)
#endif

// 16-byte global->LDS copy: async DMA when available, else direct.
__device__ __forceinline__ void cp_b128(const bf16* g, bf16* l) {
#if ASYNC_LDS
  __builtin_amdgcn_global_load_async_to_lds_b128(
      (__attribute__((address_space(1))) v4i*)(bf16*)g,
      (__attribute__((address_space(3))) v4i*)l, 0, 0);
#else
  *(uint4*)l = *(const uint4*)g;
#endif
}

__device__ __forceinline__ v8f wmma_bf16(v16bf a, v16bf b, v8f c) {
  return __builtin_amdgcn_wmma_f32_16x16x32_bf16(false, a, false, b,
                                                 (short)0, c, false, false);
}

// A fragment: 16x32 bf16, row-major source with stride ld.
// K(r,half) = half*16 + 2*r   (contiguous k order, matches load_b_frag_t)
__device__ __forceinline__ v16bf load_a_frag(const bf16* p, int ld) {
  int lane = threadIdx.x & 31;
  int hf = lane >> 4, m = lane & 15;
  v16bf a;
#pragma unroll
  for (int r = 0; r < 8; ++r) {
    int k = (hf << 4) + (r << 1);
    a[2 * r]     = p[m * ld + k];
    a[2 * r + 1] = p[m * ld + k + 1];
  }
  return a;
}

// B fragment: 32x16 bf16 (KxN), source TRANSPOSED in LDS as [N][K], stride ld.
// 16 contiguous bf16 per lane -> compiler emits 2x ds_load_b128.
__device__ __forceinline__ v16bf load_b_frag_t(const bf16* p, int ld) {
  int lane = threadIdx.x & 31;
  int hf = lane >> 4, n = lane & 15;
  v16bf b;
#pragma unroll
  for (int r = 0; r < 8; ++r) {
    int k = (hf << 4) + (r << 1);
    b[2 * r]     = p[n * ld + k];
    b[2 * r + 1] = p[n * ld + k + 1];
  }
  return b;
}

// ---------------------------------------------------------------------------
// Kernel 1: qkv = x @ w_qkv  -> Q(scaled)/K bf16 [b,h,n,d]; V TRANSPOSED as
// vt[b,h,d,n] bf16.  Block tile 128(M) x 64(N), K-step 32, 8 waves.
// ---------------------------------------------------------------------------
__global__ __launch_bounds__(256) void qkv_kernel(
    const float* __restrict__ x, const float* __restrict__ w_qkv,
    bf16* __restrict__ qo, bf16* __restrict__ ko, bf16* __restrict__ vto) {
  __shared__ bf16 lds_a[128 * 32];
  __shared__ bf16 lds_b[64 * 32];
  int tid = threadIdx.x, lane = tid & 31, wid = tid >> 5;
  int row0 = blockIdx.x * 128, n0 = blockIdx.y * 64;

  v8f vz = {0.f, 0.f, 0.f, 0.f, 0.f, 0.f, 0.f, 0.f};
  v8f acc[4] = {vz, vz, vz, vz};

  for (int k0 = 0; k0 < 512; k0 += 32) {
    // x tile [128][32] fp32 -> bf16 ; 4 float4 per thread, fixed trip count
#pragma unroll
    for (int it = 0; it < 4; ++it) {
      int e = tid + it * 256;
      int i = e >> 3, qd = e & 7;
      float4 f = *(const float4*)(x + (size_t)(row0 + i) * 512 + k0 + qd * 4);
      bf16* d = lds_a + i * 32 + qd * 4;
      d[0] = (bf16)f.x; d[1] = (bf16)f.y; d[2] = (bf16)f.z; d[3] = (bf16)f.w;
    }
    // w tile transposed [n][k] ; 8 scalars per thread, fixed trip count
#pragma unroll
    for (int it = 0; it < 8; ++it) {
      int e = tid + it * 256;
      int kk = e >> 6, nn = e & 63;
      lds_b[nn * 32 + kk] = (bf16)w_qkv[(size_t)(k0 + kk) * 1536 + n0 + nn];
    }
    __syncthreads();
    if (k0 + 32 < 512) {  // prefetch next tiles -> global_prefetch_b8
      __builtin_prefetch(x + (size_t)(row0 + (tid >> 1)) * 512 + k0 + 32 + (tid & 1) * 16, 0, 3);
      __builtin_prefetch(w_qkv + (size_t)(k0 + 32 + (tid >> 3)) * 1536 + n0 + (tid & 7) * 8, 0, 3);
    }
    v16bf a = load_a_frag(lds_a + wid * 16 * 32, 32);
#pragma unroll
    for (int t = 0; t < 4; ++t) {
      v16bf b = load_b_frag_t(lds_b + t * 16 * 32, 32);
      acc[t] = wmma_bf16(a, b, acc[t]);
    }
    __syncthreads();
  }

  // epilogue: whole block maps to one (q|k|v, head) since TN==DHEAD==64
  int hf = lane >> 4, nl = lane & 15;
  int seg = n0 / 512;               // 0=Q 1=K 2=V
  int h = (n0 % 512) / 64;
#pragma unroll
  for (int t = 0; t < 4; ++t)
#pragma unroll
    for (int r = 0; r < 8; ++r) {
      int row = row0 + wid * 16 + (hf << 3) + r;
      int bi = row >> 12, ns = row & 4095;
      int d = t * 16 + nl;
      size_t bh = (size_t)(bi * 8 + h);
      float val = acc[t][r];
      if (seg == 2) {
        vto[(bh * DHEAD + d) * SEQ + ns] = (bf16)val;          // transposed
      } else {
        bf16* dst = seg ? ko : qo;
        dst[(bh * SEQ + ns) * DHEAD + d] =
            (bf16)(val * (seg ? 1.0f : QK_SCALE));
      }
    }
}

// ---------------------------------------------------------------------------
// Kernel 2: vsum[bh][d] = sum_n vt[bh][d][n]   (contiguous reduction)
// ---------------------------------------------------------------------------
__global__ __launch_bounds__(64) void vsum_kernel(const bf16* __restrict__ vt,
                                                  float* __restrict__ vsum) {
  int bh = blockIdx.x, d = threadIdx.x;
  const bf16* p = vt + ((size_t)bh * DHEAD + d) * SEQ;
  float s = 0.f;
  for (int n = 0; n < SEQ; ++n) s += (float)p[n];
  vsum[bh * DHEAD + d] = s;
}

// ---------------------------------------------------------------------------
// Kernel 3: streaming attention.  grid = NBH * (SEQ/128), 8 waves x 16 rows.
// 128 query rows share each K/V^T tile (halves K/V traffic vs 4-wave blocks).
// T += S_tile @ V_tile (no softmax rescale), online logsumexp per row,
// out = T - lse * vsum.  Double-buffered async tiles: 1 K + 1 V b128/thread.
// ---------------------------------------------------------------------------
__global__ __launch_bounds__(256) void attn_kernel(
    const bf16* __restrict__ q, const bf16* __restrict__ kx,
    const bf16* __restrict__ vt, const float* __restrict__ vsum,
    bf16* __restrict__ ao) {
  __shared__ bf16 ksm[2][32 * 64];    // K tile, row-major [j][d]
  __shared__ bf16 vtm[2][64 * 32];    // V^T tile [d][j]
  __shared__ bf16 ssm[8][16 * 32];    // per-wave S staging (C->A round trip)

  int tid = threadIdx.x, lane = tid & 31, wid = tid >> 5;
  int bh = blockIdx.x >> 5, iblk = blockIdx.x & 31;
  int i0 = iblk * 128 + wid * 16;
  int hf = lane >> 4, nl = lane & 15;

  // Q fragments: loaded once, reused across the whole j-loop.
  const bf16* qp = q + ((size_t)bh * SEQ + i0) * DHEAD;
  v16bf aq0 = load_a_frag(qp, DHEAD);
  v16bf aq1 = load_a_frag(qp + 32, DHEAD);

  const bf16* kbase = kx + (size_t)bh * SEQ * DHEAD;
  const bf16* vbase = vt + (size_t)bh * DHEAD * SEQ;

  // exactly one K b128 + one V^T b128 per thread per tile (256 threads)
  int dd = tid >> 2, qq = tid & 3;
  auto load_tiles = [&](int jt, int buf) {
    cp_b128(kbase + (size_t)jt * 32 * DHEAD + tid * 8, &ksm[buf][tid * 8]);
    cp_b128(vbase + (size_t)dd * SEQ + jt * 32 + qq * 8,
            &vtm[buf][dd * 32 + qq * 8]);
  };

  v8f vz = {0.f, 0.f, 0.f, 0.f, 0.f, 0.f, 0.f, 0.f};
  v8f T[4] = {vz, vz, vz, vz};
  float mrun[8], srun[8];
#pragma unroll
  for (int r = 0; r < 8; ++r) { mrun[r] = -1e30f; srun[r] = 0.f; }

  const int NJ = SEQ / 32;
  load_tiles(0, 0);
  for (int jt = 0; jt < NJ; ++jt) {
    int buf = jt & 1;
    if (jt + 1 < NJ) {
      load_tiles(jt + 1, buf ^ 1);
      WAIT_ASYNC(2);     // retire current buffer's 2 loads (in-order)
    } else {
      WAIT_ASYNC(0);
    }
    __syncthreads();

    // S[16 x 32] = Qs (16x64) @ K^T : contraction dim d (two k=32 chunks)
    v8f S[2];
#pragma unroll
    for (int j2 = 0; j2 < 2; ++j2) {
      v16bf b0 = load_b_frag_t(&ksm[buf][j2 * 16 * 64], 64);       // d 0..31
      v16bf b1 = load_b_frag_t(&ksm[buf][j2 * 16 * 64 + 32], 64);  // d 32..63
      S[j2] = wmma_bf16(aq0, b0, vz);
      S[j2] = wmma_bf16(aq1, b1, S[j2]);
    }

    // Online logsumexp + stage S (bf16) for the A-side of S@V.
#pragma unroll
    for (int r = 0; r < 8; ++r) {
      int m = (hf << 3) + r;
      float s0 = S[0][r], s1 = S[1][r];
      float nm = fmaxf(mrun[r], fmaxf(s0, s1));
      srun[r] = srun[r] * __expf(mrun[r] - nm) + __expf(s0 - nm) + __expf(s1 - nm);
      mrun[r] = nm;
      ssm[wid][m * 32 + nl]      = (bf16)s0;
      ssm[wid][m * 32 + 16 + nl] = (bf16)s1;
    }
    // Per-wave LDS RAW: DS ops are in-order within a wave.
    v16bf as = load_a_frag(ssm[wid], 32);
#pragma unroll
    for (int dt = 0; dt < 4; ++dt) {
      v16bf bv = load_b_frag_t(&vtm[buf][dt * 16 * 32], 32);
      T[dt] = wmma_bf16(as, bv, T[dt]);
    }
    __syncthreads();
  }

  // lse: combine (m, sumexp) across the 16 lanes holding one row's columns.
  float lse[8];
#pragma unroll
  for (int r = 0; r < 8; ++r) {
    float m = mrun[r], s = srun[r];
#pragma unroll
    for (int mask = 1; mask < 16; mask <<= 1) {
      float om = __shfl_xor(m, mask, 32);
      float os = __shfl_xor(s, mask, 32);
      float nm = fmaxf(m, om);
      s = s * __expf(m - nm) + os * __expf(om - nm);
      m = nm;
    }
    lse[r] = m + __logf(s);
  }

  int bi = bh >> 3, h = bh & 7;
#pragma unroll
  for (int dt = 0; dt < 4; ++dt) {
    float vs = vsum[bh * DHEAD + dt * 16 + nl];
#pragma unroll
    for (int r = 0; r < 8; ++r) {
      int ns = i0 + (hf << 3) + r;
      float val = T[dt][r] - lse[r] * vs;
      ao[((size_t)bi * SEQ + ns) * 512 + h * 64 + dt * 16 + nl] = (bf16)val;
    }
  }
}

// ---------------------------------------------------------------------------
// Kernel 4: out = attn_out @ w_out + b_out + x   (fp32 out, residual fused)
// ---------------------------------------------------------------------------
__global__ __launch_bounds__(256) void proj_kernel(
    const bf16* __restrict__ ao, const float* __restrict__ w_out,
    const float* __restrict__ b_out, const float* __restrict__ x,
    float* __restrict__ out) {
  __shared__ bf16 lds_a[128 * 32];
  __shared__ bf16 lds_b[64 * 32];
  int tid = threadIdx.x, lane = tid & 31, wid = tid >> 5;
  int row0 = blockIdx.x * 128, n0 = blockIdx.y * 64;

  v8f vz = {0.f, 0.f, 0.f, 0.f, 0.f, 0.f, 0.f, 0.f};
  v8f acc[4] = {vz, vz, vz, vz};

  for (int k0 = 0; k0 < 512; k0 += 32) {
#pragma unroll
    for (int it = 0; it < 2; ++it) {   // A already bf16: 2 b128 per thread
      int e = tid + it * 256;
      int i = e >> 2, qd = e & 3;
      ((uint4*)lds_a)[i * 4 + qd] =
          *(const uint4*)(ao + (size_t)(row0 + i) * 512 + k0 + qd * 8);
    }
#pragma unroll
    for (int it = 0; it < 8; ++it) {
      int e = tid + it * 256;
      int kk = e >> 6, nn = e & 63;
      lds_b[nn * 32 + kk] = (bf16)w_out[(size_t)(k0 + kk) * 512 + n0 + nn];
    }
    __syncthreads();
    if (k0 + 32 < 512) {
      __builtin_prefetch(ao + (size_t)(row0 + (tid >> 1)) * 512 + k0 + 32 + (tid & 1) * 16, 0, 3);
      __builtin_prefetch(w_out + (size_t)(k0 + 32 + (tid >> 3)) * 512 + n0 + (tid & 7) * 8, 0, 3);
    }
    v16bf a = load_a_frag(lds_a + wid * 16 * 32, 32);
#pragma unroll
    for (int t = 0; t < 4; ++t) {
      v16bf b = load_b_frag_t(lds_b + t * 16 * 32, 32);
      acc[t] = wmma_bf16(a, b, acc[t]);
    }
    __syncthreads();
  }

  int hf = lane >> 4, nl = lane & 15;
#pragma unroll
  for (int t = 0; t < 4; ++t)
#pragma unroll
    for (int r = 0; r < 8; ++r) {
      size_t row = (size_t)row0 + wid * 16 + (hf << 3) + r;
      int col = n0 + t * 16 + nl;
      out[row * 512 + col] = acc[t][r] + b_out[col] + x[row * 512 + col];
    }
}

// ---------------------------------------------------------------------------
extern "C" void kernel_launch(void* const* d_in, const int* in_sizes, int n_in,
                              void* d_out, int out_size, void* d_ws, size_t ws_size,
                              hipStream_t stream) {
  const float* x     = (const float*)d_in[0];
  const float* w_qkv = (const float*)d_in[1];
  const float* w_out = (const float*)d_in[2];
  const float* b_out = (const float*)d_in[3];
  float* out = (float*)d_out;

  // workspace: Q, K, V^T, attn_out bf16 (4 x 8MB) + vsum fp32 (4KB)
  const size_t E = (size_t)2 * 8 * SEQ * DHEAD;   // 4,194,304 elems
  bf16* q  = (bf16*)d_ws;
  bf16* k  = q + E;
  bf16* vt = k + E;
  bf16* ao = vt + E;
  float* vsum = (float*)(ao + E);

  qkv_kernel <<<dim3(64, 24), 256, 0, stream>>>(x, w_qkv, q, k, vt);
  vsum_kernel<<<NBH, DHEAD, 0, stream>>>(vt, vsum);
  attn_kernel<<<NBH * (SEQ / 128), 256, 0, stream>>>(q, k, vt, vsum, ao);
  proj_kernel<<<dim3(64, 8), 256, 0, stream>>>(ao, w_out, b_out, x, out);
}
